// LSTM_44006234915041
// MI455X (gfx1250) — compile-verified, matrix operand order checked by reading, not run on hardware
//
#include <hip/hip_runtime.h>
#include <hip/hip_bf16.h>
#include <math.h>

// ---------------------------------------------------------------------------
// Problem constants (from reference)
// ---------------------------------------------------------------------------
#define BB    32768   // batch
#define TT    25      // time steps
#define DIN   4       // input features
#define HH    32      // LSTM hidden
#define G4H   128     // 4*H gate width
#define HIDD  64      // MLP hidden
#define NBLK  5       // number of LSTM blocks

typedef _Float16 v16h __attribute__((ext_vector_type(16)));
typedef float    v8f  __attribute__((ext_vector_type(8)));
typedef unsigned int u32x4 __attribute__((ext_vector_type(4)));

union V16H {
    v16h     v;
    u32x4    u[2];
    _Float16 h[16];
};

__device__ __forceinline__ v8f splat8(float x) {
    return (v8f){x, x, x, x, x, x, x, x};
}

// Hardware transcendental tanh (CDNA5 TRANS pipe, co-executes with VALU).
// ISA 7.4: one independent op or V_NOP required after a TRANS op before its
// result is consumed -- the compiler can't see through the asm, so pad here.
__device__ __forceinline__ float fast_tanh(float x) {
    float r;
    asm volatile("v_tanh_f32 %0, %1\n\tv_nop" : "=v"(r) : "v"(x));
    return r;
}
// sigmoid(x) = 0.5*tanh(x/2) + 0.5
__device__ __forceinline__ float fast_sigm(float x) {
    return fmaf(0.5f, fast_tanh(0.5f * x), 0.5f);
}

// K base for VGPR-pair q (0..7) in the CDNA5 16-bit A/B operand layout.
// lane<16: K = {0..7, 16..23}; lane>=16: K = {8..15, 24..31}, 2 halfs/VGPR.
__device__ __forceinline__ int kbase(int c0, int q) {
    return (q < 4) ? (c0 + 2 * q) : (c0 + 16 + 2 * (q - 4));
}

// Load a 16-half B/A fragment stored contiguously (32B per lane) from LDS.
__device__ __forceinline__ v16h load_frag32(const _Float16* p) {
    V16H f;
    f.u[0] = *(const u32x4*)(p);
    f.u[1] = *(const u32x4*)(p + 8);
    return f.v;
}

#define WMMA_F16(A, B, C) \
    __builtin_amdgcn_wmma_f32_16x16x32_f16(false, (A), false, (B), (short)0, (C), false, false)

// ---------------------------------------------------------------------------
// Kernel 1: fused 2-layer LSTM per block, wave = 16 batch rows.
// grid = (B/64, NB), block = 128 (4 waves, each an independent batch tile).
// Writes hid[b][blk][32] (f32) to workspace.
// ---------------------------------------------------------------------------
__global__ void __launch_bounds__(128)
lstm_blocks_kernel(const float* __restrict__ x,    const float* __restrict__ mask,
                   const float* __restrict__ Wih0, const float* __restrict__ Whh0,
                   const float* __restrict__ bih0, const float* __restrict__ bhh0,
                   const float* __restrict__ Wih1, const float* __restrict__ Whh1,
                   const float* __restrict__ bih1, const float* __restrict__ bhh1,
                   float* __restrict__ hid)
{
    // Weight fragments in B-operand layout, f16: [matrix][n-tile][lane][16 halfs]
    // matrices: 0 = Wih0 (K padded 4->32), 1 = Whh0, 2 = Wih1, 3 = Whh1
    __shared__ _Float16 wfrag[4][8][32][16];       // 32 KB
    __shared__ _Float16 h0buf[4][16][32];          // per-wave h exchange, 4 KB
    __shared__ _Float16 h1buf[4][16][32];          // 4 KB

    const int tid  = threadIdx.x;
    const int lane = tid & 31;
    const int w    = tid >> 5;          // wave id in WG
    const int blk  = blockIdx.y;        // LSTM block index 0..4
    const int colL = lane & 15;
    const int c0   = (lane & 16) ? 8 : 0;
    const int hi8  = (lane & 16) ? 8 : 0;
    const int mrow = lane & 15;

    // --- build this wave's weight matrix fragments (one matrix per wave) ---
    {
        const float* Wsrc;
        int Kld, Kvalid;
        if (w == 0)      { Wsrc = Wih0 + (size_t)blk * G4H * DIN; Kld = DIN; Kvalid = DIN; }
        else if (w == 1) { Wsrc = Whh0 + (size_t)blk * G4H * HH;  Kld = HH;  Kvalid = HH;  }
        else if (w == 2) { Wsrc = Wih1 + (size_t)blk * G4H * HH;  Kld = HH;  Kvalid = HH;  }
        else             { Wsrc = Whh1 + (size_t)blk * G4H * HH;  Kld = HH;  Kvalid = HH;  }
        for (int n = 0; n < 8; ++n) {
            const int col = 16 * n + colL;
            #pragma unroll
            for (int q = 0; q < 8; ++q) {
                const int K0 = kbase(c0, q);
                #pragma unroll
                for (int e = 0; e < 2; ++e) {
                    const int K = K0 + e;
                    float v = (K < Kvalid) ? Wsrc[col * Kld + K] : 0.0f;
                    wfrag[w][n][lane][2 * q + e] = (_Float16)v;
                }
            }
        }
    }

    // --- per-lane biases (b = bih + bhh), broadcast along M ---
    float b0r[8], b1r[8];
    #pragma unroll
    for (int n = 0; n < 8; ++n) {
        const int col = 16 * n + colL;
        b0r[n] = bih0[blk * G4H + col] + bhh0[blk * G4H + col];
        b1r[n] = bih1[blk * G4H + col] + bhh1[blk * G4H + col];
    }

    // --- zero the per-wave hidden buffers ---
    {
        _Float16* p0 = &h0buf[w][0][0];
        _Float16* p1 = &h1buf[w][0][0];
        for (int idx = lane; idx < 16 * 32; idx += 32) {
            p0[idx] = (_Float16)0.0f;
            p1[idx] = (_Float16)0.0f;
        }
    }
    __syncthreads();

    const int rbase = (blockIdx.x * 4 + w) * 16;   // first batch row of this wave
    const int Ti = (blk + 1) * 5;
    const int t0 = TT - Ti;

    v8f c0a[2], c1a[2];
    c0a[0] = splat8(0.0f); c0a[1] = splat8(0.0f);
    c1a[0] = splat8(0.0f); c1a[1] = splat8(0.0f);

    for (int t = t0; t < TT; ++t) {
        // ---- x A-fragment (K=0..3 real, rest zero) ----
        V16H xa;
        #pragma unroll
        for (int j = 0; j < 16; ++j) xa.h[j] = (_Float16)0.0f;
        if (lane < 16) {
            const int b = rbase + lane;
            const float4 xv = *(const float4*)(x    + ((size_t)b * TT + t) * DIN);
            const float4 mv = *(const float4*)(mask + ((size_t)b * TT + t) * DIN);
            xa.h[0] = (_Float16)(xv.x * mv.x);
            xa.h[1] = (_Float16)(xv.y * mv.y);
            xa.h[2] = (_Float16)(xv.z * mv.z);
            xa.h[3] = (_Float16)(xv.w * mv.w);
        }

        // ---- recurrent h fragments (A layout: 2 x b128 per lane) ----
        V16H h0a, h1a;
        h0a.u[0] = *(const u32x4*)&h0buf[w][mrow][c0];
        h0a.u[1] = *(const u32x4*)&h0buf[w][mrow][c0 + 16];
        h1a.u[0] = *(const u32x4*)&h1buf[w][mrow][c0];
        h1a.u[1] = *(const u32x4*)&h1buf[w][mrow][c0 + 16];

        // ---- layer 0: process column halves t2=0,1 with 4 live accumulators
        //      tiles (t2, t2+2, t2+4, t2+6) == (i, f, g, o) for cols 16*t2+colL
        #pragma unroll
        for (int t2 = 0; t2 < 2; ++t2) {
            v8f acc[4];
            #pragma unroll
            for (int g = 0; g < 4; ++g) acc[g] = splat8(b0r[t2 + 2 * g]);
            #pragma unroll
            for (int g = 0; g < 4; ++g)
                acc[g] = WMMA_F16(xa.v,  load_frag32(&wfrag[0][t2 + 2 * g][lane][0]), acc[g]);
            #pragma unroll
            for (int g = 0; g < 4; ++g)
                acc[g] = WMMA_F16(h0a.v, load_frag32(&wfrag[1][t2 + 2 * g][lane][0]), acc[g]);
            #pragma unroll
            for (int v = 0; v < 8; ++v) {
                const float iv = fast_sigm(acc[0][v]);
                const float fv = fast_sigm(acc[1][v]);
                const float gv = fast_tanh(acc[2][v]);
                const float ov = fast_sigm(acc[3][v]);
                const float cc = fmaf(fv, c0a[t2][v], iv * gv);
                c0a[t2][v] = cc;
                const float hv = ov * fast_tanh(cc);
                h0buf[w][v + hi8][16 * t2 + colL] = (_Float16)hv;
            }
        }
        __syncthreads();

        h0a.u[0] = *(const u32x4*)&h0buf[w][mrow][c0];
        h0a.u[1] = *(const u32x4*)&h0buf[w][mrow][c0 + 16];

        // ---- layer 1: same two-half scheme ----
        #pragma unroll
        for (int t2 = 0; t2 < 2; ++t2) {
            v8f acc[4];
            #pragma unroll
            for (int g = 0; g < 4; ++g) acc[g] = splat8(b1r[t2 + 2 * g]);
            #pragma unroll
            for (int g = 0; g < 4; ++g)
                acc[g] = WMMA_F16(h0a.v, load_frag32(&wfrag[2][t2 + 2 * g][lane][0]), acc[g]);
            #pragma unroll
            for (int g = 0; g < 4; ++g)
                acc[g] = WMMA_F16(h1a.v, load_frag32(&wfrag[3][t2 + 2 * g][lane][0]), acc[g]);
            #pragma unroll
            for (int v = 0; v < 8; ++v) {
                const float iv = fast_sigm(acc[0][v]);
                const float fv = fast_sigm(acc[1][v]);
                const float gv = fast_tanh(acc[2][v]);
                const float ov = fast_sigm(acc[3][v]);
                const float cc = fmaf(fv, c1a[t2][v], iv * gv);
                c1a[t2][v] = cc;
                const float hv = ov * fast_tanh(cc);
                h1buf[w][v + hi8][16 * t2 + colL] = (_Float16)hv;
                if (t == TT - 1) {
                    const size_t row = (size_t)(rbase + v + hi8);
                    hid[(row * NBLK + blk) * HH + 16 * t2 + colL] = hv;
                }
            }
        }
        __syncthreads();
    }
}

// ---------------------------------------------------------------------------
// Kernel 2: MLP head on (B*NB, 32) rows -> GELU -> (B*NB, 64).
// grid = B*NB/64, block = 128 (4 waves x 16 rows).
// ---------------------------------------------------------------------------
__global__ void __launch_bounds__(128)
mlp_head_kernel(const float* __restrict__ hid,
                const float* __restrict__ W1, const float* __restrict__ b1,
                const float* __restrict__ W2, const float* __restrict__ b2,
                float* __restrict__ out)
{
    __shared__ _Float16 w1f[4][32][16];            // W1^T fragments (K=32, 4 N-tiles)
    __shared__ _Float16 w2f[2][4][32][16];         // W2^T fragments (2 K-halves x 4 N-tiles)
    __shared__ _Float16 hbuf[4][16][HIDD];         // per-wave GELU output exchange

    const int tid  = threadIdx.x;
    const int lane = tid & 31;
    const int w    = tid >> 5;
    const int colL = lane & 15;
    const int c0   = (lane & 16) ? 8 : 0;
    const int hi8  = (lane & 16) ? 8 : 0;
    const int mrow = lane & 15;

    // --- build fragments: wave w -> W1 tile w; W2 combos w and w+4 ---
    {
        const int n = w;
        #pragma unroll
        for (int q = 0; q < 8; ++q) {
            const int K0 = kbase(c0, q);
            #pragma unroll
            for (int e = 0; e < 2; ++e)
                w1f[n][lane][2 * q + e] = (_Float16)W1[(16 * n + colL) * HH + K0 + e];
        }
    }
    #pragma unroll
    for (int rep = 0; rep < 2; ++rep) {
        const int idx = w + 4 * rep;
        const int kh  = idx >> 2;
        const int n   = idx & 3;
        #pragma unroll
        for (int q = 0; q < 8; ++q) {
            const int K0 = kbase(c0, q);
            #pragma unroll
            for (int e = 0; e < 2; ++e)
                w2f[kh][n][lane][2 * q + e] =
                    (_Float16)W2[(16 * n + colL) * HIDD + 32 * kh + K0 + e];
        }
    }
    float b1r[4], b2r[4];
    #pragma unroll
    for (int n = 0; n < 4; ++n) {
        b1r[n] = b1[16 * n + colL];
        b2r[n] = b2[16 * n + colL];
    }
    __syncthreads();

    const int rbase = (blockIdx.x * 4 + w) * 16;   // row in flattened (B*NB)

    // --- load hidden row A-fragment (f32 global -> f16) ---
    V16H ha;
    {
        const float* hp = hid + (size_t)(rbase + mrow) * HH;
        #pragma unroll
        for (int j = 0; j < 8; ++j) ha.h[j]     = (_Float16)hp[c0 + j];
        #pragma unroll
        for (int j = 0; j < 8; ++j) ha.h[8 + j] = (_Float16)hp[c0 + 16 + j];
    }

    // --- layer 1: hidden @ W1^T + b1, exact GELU ---
    v8f acc[4];
    #pragma unroll
    for (int n = 0; n < 4; ++n) {
        acc[n] = splat8(b1r[n]);
        acc[n] = WMMA_F16(ha.v, load_frag32(&w1f[n][lane][0]), acc[n]);
    }
    #pragma unroll
    for (int n = 0; n < 4; ++n) {
        #pragma unroll
        for (int v = 0; v < 8; ++v) {
            const float xv = acc[n][v];
            const float g  = 0.5f * xv * (1.0f + erff(xv * 0.70710678118654752f));
            hbuf[w][v + hi8][16 * n + colL] = (_Float16)g;
        }
    }
    __syncthreads();

    // --- layer 2: h @ W2^T + b2 (K=64 = two chained K=32 WMMAs) ---
    V16H a0, a1;
    a0.u[0] = *(const u32x4*)&hbuf[w][mrow][c0];
    a0.u[1] = *(const u32x4*)&hbuf[w][mrow][c0 + 16];
    a1.u[0] = *(const u32x4*)&hbuf[w][mrow][c0 + 32];
    a1.u[1] = *(const u32x4*)&hbuf[w][mrow][c0 + 48];

    #pragma unroll
    for (int n = 0; n < 4; ++n) {
        v8f o = splat8(b2r[n]);
        o = WMMA_F16(a0.v, load_frag32(&w2f[0][n][lane][0]), o);
        o = WMMA_F16(a1.v, load_frag32(&w2f[1][n][lane][0]), o);
        #pragma unroll
        for (int v = 0; v < 8; ++v)
            out[(size_t)(rbase + v + hi8) * HIDD + 16 * n + colL] = o[v];
    }
}

// ---------------------------------------------------------------------------
extern "C" void kernel_launch(void* const* d_in, const int* in_sizes, int n_in,
                              void* d_out, int out_size, void* d_ws, size_t ws_size,
                              hipStream_t stream)
{
    const float* x    = (const float*)d_in[0];
    const float* mask = (const float*)d_in[1];
    const float* Wih0 = (const float*)d_in[2];
    const float* Whh0 = (const float*)d_in[3];
    const float* bih0 = (const float*)d_in[4];
    const float* bhh0 = (const float*)d_in[5];
    const float* Wih1 = (const float*)d_in[6];
    const float* Whh1 = (const float*)d_in[7];
    const float* bih1 = (const float*)d_in[8];
    const float* bhh1 = (const float*)d_in[9];
    const float* W1   = (const float*)d_in[10];
    const float* b1   = (const float*)d_in[11];
    const float* W2   = (const float*)d_in[12];
    const float* b2   = (const float*)d_in[13];

    float* hid = (float*)d_ws;   // [B][NB][32] f32 = 20 MB
    float* out = (float*)d_out;  // [B][NB][64] f32

    dim3 g1(BB / 64, NBLK);
    lstm_blocks_kernel<<<g1, 128, 0, stream>>>(x, mask, Wih0, Whh0, bih0, bhh0,
                                               Wih1, Whh1, bih1, bhh1, hid);

    dim3 g2((BB * NBLK) / 64);
    mlp_head_kernel<<<g2, 128, 0, stream>>>(hid, W1, b1, W2, b2, out);
}
